// MultiHeadedAttention_45887430591273
// MI455X (gfx1250) — matrix operand order repaired and checked by reading, not compiled, not run on hardware
//
#include <hip/hip_runtime.h>
#include <hip/hip_bf16.h>

// ---------------------------------------------------------------------------
// MHA for MI455X (gfx1250, wave32, WMMA).  N=4, L=2048, D=512, H=8, DK=64.
// All matmuls via v_wmma_f32_16x16x32_f16 (f16 in, fp32 accumulate).
// K-block staging uses GLOBAL_LOAD_ASYNC_TO_LDS_B128 (ASYNCcnt) and the key
// loop prefetches the next block (global_prefetch_b8).
// ---------------------------------------------------------------------------

typedef __attribute__((ext_vector_type(16))) _Float16 v16h;
typedef __attribute__((ext_vector_type(8)))  _Float16 v8h;
typedef __attribute__((ext_vector_type(8)))  float    v8f;
typedef __attribute__((ext_vector_type(4)))  float    v4f;

#define HN     8
#define DMODEL 512
#define DKH    64
#define NBATCH 4
#define LSEQ   2048

__device__ __forceinline__ v8f wmma_f16(v16h a, v16h b, v8f c) {
  return __builtin_amdgcn_wmma_f32_16x16x32_f16(false, a, false, b,
                                                (short)0, c, false, false);
}

// Async copy 16B global -> LDS (no VGPR round trip; tracked by ASYNCcnt).
// Generic shared pointers are {aperture_hi32, lds_offset32}, so the low 32
// bits are the workgroup-relative LDS address the instruction wants.
__device__ __forceinline__ void async_ld_b128(void* lds, const void* g) {
  unsigned lds_off = (unsigned)(size_t)lds;
  unsigned long long ga = (unsigned long long)(size_t)g;
  asm volatile("global_load_async_to_lds_b128 %0, %1, off"
               :: "v"(lds_off), "v"(ga)
               : "memory");
}
__device__ __forceinline__ void wait_async0() {
  asm volatile("s_wait_asynccnt 0x0" ::: "memory");
}

// Build a 16-half fragment from two 16-byte chunks (-> ds_load_b128 pairs).
__device__ __forceinline__ v16h ld_frag16(const _Float16* p0, const _Float16* p1) {
  v8h lo = *(const v8h*)p0;
  v8h hi = *(const v8h*)p1;
  v16h r;
#pragma unroll
  for (int i = 0; i < 8; ++i) { r[i] = lo[i]; r[i + 8] = hi[i]; }
  return r;
}

// Stage 16 contiguous values (fp32 or f16 source) into LDS as f16.
__device__ __forceinline__ void stage16(const float* g, _Float16* lds) {
  const v4f* gv = (const v4f*)g;
  v4f f0 = gv[0], f1 = gv[1], f2 = gv[2], f3 = gv[3];
  v8h h0, h1;
#pragma unroll
  for (int i = 0; i < 4; ++i) {
    h0[i]     = (_Float16)f0[i];
    h0[4 + i] = (_Float16)f1[i];
    h1[i]     = (_Float16)f2[i];
    h1[4 + i] = (_Float16)f3[i];
  }
  v8h* d = (v8h*)lds;
  d[0] = h0; d[1] = h1;
}
__device__ __forceinline__ void stage16(const _Float16* g, _Float16* lds) {
  const v8h* gv = (const v8h*)g;
  v8h a = gv[0], b = gv[1];
  v8h* d = (v8h*)lds;
  d[0] = a; d[1] = b;
}
__device__ __forceinline__ void stage8(const float* g, _Float16* lds) {
  const v4f* gv = (const v4f*)g;
  v4f f0 = gv[0], f1 = gv[1];
  v8h h0;
#pragma unroll
  for (int i = 0; i < 4; ++i) { h0[i] = (_Float16)f0[i]; h0[4 + i] = (_Float16)f1[i]; }
  *(v8h*)lds = h0;
}

// ---------------------------------------------------------------------------
// Y = X @ W^T + bias.   X: 8192 x 512 (TIn), W: 512 x 512 fp32 (rows = out
// features, contiguous along K), bias: 512 fp32.
//   OUTF32=false: Y f16, head-major (n,h,l,dk), scaled by ascale (Q: 0.125)
//   OUTF32=true : Y fp32 row-major (final projection into d_out)
// Block: 256 thr = 8 waves; tile 128(M) x 64(N), K stepped by 32.
// ---------------------------------------------------------------------------
template <typename TIn, bool OUTF32>
__global__ __launch_bounds__(256) void gemm_xwt(const TIn* __restrict__ X,
                                                const float* __restrict__ W,
                                                const float* __restrict__ bias,
                                                void* __restrict__ Yv,
                                                float ascale) {
  __shared__ _Float16 As[128][40];  // +8 halves pad, rows 16B-aligned
  __shared__ _Float16 Bs[64][40];

  const int tid  = threadIdx.x;
  const int lane = tid & 31;
  const int w    = tid >> 5;
  const int wm   = w >> 1, wn = w & 1;    // 4x2 wave grid -> 32x32 per wave
  const int l16  = lane & 15;
  const int hi   = lane >> 4;             // lane half selects K-chunk phase
  const int bm   = blockIdx.x, bn = blockIdx.y;

  // Loop-invariant LDS fragment base pointers (documented WMMA layouts).
  const _Float16* pa[2];
  const _Float16* pb[2];
#pragma unroll
  for (int mt = 0; mt < 2; ++mt) pa[mt] = &As[wm * 32 + mt * 16 + l16][hi ? 8 : 0];
#pragma unroll
  for (int nt = 0; nt < 2; ++nt) pb[nt] = &Bs[wn * 32 + nt * 16 + l16][hi ? 8 : 0];

  v8f acc[2][2];
#pragma unroll
  for (int mt = 0; mt < 2; ++mt)
#pragma unroll
    for (int nt = 0; nt < 2; ++nt) { v8f z = {}; acc[mt][nt] = z; }

  const int    ar = tid >> 1, ac = (tid & 1) * 16;   // A stage: 16 elems/thr
  const int    br = tid >> 2, bc = (tid & 3) * 8;    // B stage: 8 elems/thr
  const size_t Xrow = (size_t)(bm * 128 + ar) * DMODEL;
  const size_t Wrow = (size_t)(bn * 64 + br) * DMODEL;

  for (int k0 = 0; k0 < DMODEL; k0 += 32) {
    stage16(X + Xrow + k0 + ac, &As[ar][ac]);
    stage8 (W + Wrow + k0 + bc, &Bs[br][bc]);
    if (k0 + 32 < DMODEL) {   // prefetch next K-chunk into cache
      __builtin_prefetch(X + Xrow + k0 + 32 + ac, 0, 0);
      __builtin_prefetch(W + Wrow + k0 + 32 + bc, 0, 0);
    }
    __syncthreads();
#pragma unroll
    for (int mt = 0; mt < 2; ++mt) {
      v16h a = ld_frag16(pa[mt], pa[mt] + 16);
#pragma unroll
      for (int nt = 0; nt < 2; ++nt) {
        v16h b = ld_frag16(pb[nt], pb[nt] + 16);
        acc[mt][nt] = wmma_f16(a, b, acc[mt][nt]);
      }
    }
    __syncthreads();
  }

  // Epilogue: C layout = lane 0..15 -> col, VGPR v -> row v (+8 upper half).
#pragma unroll
  for (int mt = 0; mt < 2; ++mt) {
#pragma unroll
    for (int nt = 0; nt < 2; ++nt) {
      const int rbase = bm * 128 + wm * 32 + mt * 16 + hi * 8;
      const int cg    = bn * 64 + wn * 32 + nt * 16 + l16;
      const float bv  = bias[cg];
#pragma unroll
      for (int v = 0; v < 8; ++v) {
        const int r = rbase + v;
        const float val = (acc[mt][nt][v] + bv) * ascale;
        if constexpr (OUTF32) {
          ((float*)Yv)[(size_t)r * DMODEL + cg] = val;
        } else {
          const int n = r >> 11, l = r & (LSEQ - 1);
          const int h = cg >> 6, dk = cg & (DKH - 1);
          ((_Float16*)Yv)[(((size_t)(n * HN + h)) * LSEQ + l) * DKH + dk] =
              (_Float16)val;
        }
      }
    }
  }
}

// ---------------------------------------------------------------------------
// Flash attention. Q/K/V f16 head-major (bh, l, dk); Q pre-scaled by 1/8.
// grid (L/128, N*H), 256 thr; each wave owns 16 query rows, keys in blocks
// of 32 shared by all waves via LDS. K-block staged with async global->LDS
// copies; V-block staged transposed through VGPRs. Output ctx f16 (n, l, D).
// ---------------------------------------------------------------------------
__global__ __launch_bounds__(256) void attn_fa(const _Float16* __restrict__ Q,
                                               const _Float16* __restrict__ Kh,
                                               const _Float16* __restrict__ Vh,
                                               const int* __restrict__ Msk,
                                               _Float16* __restrict__ Ctx) {
  __shared__ _Float16 Ks[32][72];      // [key][d]   for S = Q.K^T  B-frags
  __shared__ _Float16 Vt[64][40];      // [d][key]   for O += P.V   B-frags
  __shared__ _Float16 Ps[8][16][40];   // per-wave P transpose scratch

  const int tid  = threadIdx.x;
  const int lane = tid & 31;
  const int w    = tid >> 5;
  const int l16  = lane & 15;
  const int hi   = lane >> 4;
  const int bh   = blockIdx.y;
  const int n    = bh >> 3, h = bh & 7;
  const int m0   = blockIdx.x * 128 + w * 16;

  const _Float16* Qb = Q  + (size_t)bh * LSEQ * DKH;
  const _Float16* Kb = Kh + (size_t)bh * LSEQ * DKH;
  const _Float16* Vb = Vh + (size_t)bh * LSEQ * DKH;
  const int*    mrow = Msk + (size_t)n * LSEQ;

  // Q A-fragments (16 rows x 64 d = two 16x32 frags), direct global b128s.
  v16h qf[2];
#pragma unroll
  for (int kd = 0; kd < 2; ++kd) {
    const _Float16* p = Qb + (size_t)(m0 + l16) * DKH + kd * 32 + (hi ? 8 : 0);
    qf[kd] = ld_frag16(p, p + 16);
  }

  float mrun[8], lrun[8];
  v8f o[4];
#pragma unroll
  for (int v = 0; v < 8; ++v) { mrun[v] = -1e30f; lrun[v] = 0.f; }
#pragma unroll
  for (int t = 0; t < 4; ++t) { v8f z = {}; o[t] = z; }

  const int key_ld = tid >> 3;          // 32 keys x 64 d / 256 thr = v8h each
  const int c_ld   = (tid & 7) * 8;

  for (int k0 = 0; k0 < LSEQ; k0 += 32) {
    __syncthreads();
    // K block: pure copy -> async global->LDS (ASYNCcnt), no VGPR round trip.
    async_ld_b128(&Ks[key_ld][c_ld],
                  Kb + (size_t)(k0 + key_ld) * DKH + c_ld);
    // V block: transpose in flight through VGPRs.
    v8h vv = *(const v8h*)(Vb + (size_t)(k0 + key_ld) * DKH + c_ld);
#pragma unroll
    for (int i = 0; i < 8; ++i) Vt[c_ld + i][key_ld] = vv[i];
    if (k0 + 32 < LSEQ) {  // prefetch next key block
      __builtin_prefetch(Kb + (size_t)(k0 + 32 + key_ld) * DKH + c_ld, 0, 0);
      __builtin_prefetch(Vb + (size_t)(k0 + 32 + key_ld) * DKH + c_ld, 0, 0);
    }
    wait_async0();          // K bytes landed in LDS before the barrier
    __syncthreads();

    // S = Q.K^T : two 16-key column tiles, K-dim (d)=64 in two steps.
    v8f s0 = {}, s1 = {};
#pragma unroll
    for (int kd = 0; kd < 2; ++kd) {
      const _Float16* p0 = &Ks[l16][kd * 32 + (hi ? 8 : 0)];
      const _Float16* p1 = &Ks[16 + l16][kd * 32 + (hi ? 8 : 0)];
      v16h b0 = ld_frag16(p0, p0 + 16);
      v16h b1 = ld_frag16(p1, p1 + 16);
      s0 = wmma_f16(qf[kd], b0, s0);
      s1 = wmma_f16(qf[kd], b1, s1);
    }

    // Mask (per key column; reference: score -> -1e9 where mask==0).
    const int mv0 = mrow[k0 + l16];
    const int mv1 = mrow[k0 + 16 + l16];
#pragma unroll
    for (int v = 0; v < 8; ++v) {
      if (mv0 == 0) s0[v] = -1e9f;
      if (mv1 == 0) s1[v] = -1e9f;
    }

    // Online softmax: row stats live per lane-half (row = v + 8*hi).
    float corr[8];
#pragma unroll
    for (int v = 0; v < 8; ++v) {
      float t = fmaxf(s0[v], s1[v]);
      t = fmaxf(t, __shfl_xor(t, 1));
      t = fmaxf(t, __shfl_xor(t, 2));
      t = fmaxf(t, __shfl_xor(t, 4));
      t = fmaxf(t, __shfl_xor(t, 8));
      const float mn = fmaxf(mrun[v], t);
      corr[v] = __expf(mrun[v] - mn);
      mrun[v] = mn;
    }
#pragma unroll
    for (int v = 0; v < 8; ++v) {
      s0[v] = __expf(s0[v] - mrun[v]);
      s1[v] = __expf(s1[v] - mrun[v]);
      float t = s0[v] + s1[v];
      t += __shfl_xor(t, 1);
      t += __shfl_xor(t, 2);
      t += __shfl_xor(t, 4);
      t += __shfl_xor(t, 8);
      lrun[v] = lrun[v] * corr[v] + t;
    }
#pragma unroll
    for (int t = 0; t < 4; ++t)
#pragma unroll
      for (int v = 0; v < 8; ++v) o[t][v] *= corr[v];

    // Transpose P (C layout -> A layout) through per-wave LDS scratch.
#pragma unroll
    for (int v = 0; v < 8; ++v) {
      Ps[w][v + hi * 8][l16]      = (_Float16)s0[v];
      Ps[w][v + hi * 8][16 + l16] = (_Float16)s1[v];
    }
    asm volatile("s_wait_dscnt 0x0" ::: "memory");  // per-wave LDS RAW fence
    const _Float16* pp = &Ps[w][l16][hi ? 8 : 0];
    v16h pf = ld_frag16(pp, pp + 16);

    // O += P.V : 4 column tiles over d.
#pragma unroll
    for (int nt = 0; nt < 4; ++nt) {
      const _Float16* pv = &Vt[nt * 16 + l16][hi ? 8 : 0];
      v16h bv = ld_frag16(pv, pv + 16);
      o[nt] = wmma_f16(pf, bv, o[nt]);
    }
  }

  // Normalize and store ctx row-major (n, l, h*64+dk) in f16.
#pragma unroll
  for (int v = 0; v < 8; ++v) {
    const float inv = lrun[v] > 0.f ? 1.f / lrun[v] : 0.f;
    const int r = m0 + v + hi * 8;
    const size_t rowoff = ((size_t)n * LSEQ + r) * DMODEL + h * DKH;
#pragma unroll
    for (int nt = 0; nt < 4; ++nt)
      Ctx[rowoff + nt * 16 + l16] = (_Float16)(o[nt][v] * inv);
  }
}

// ---------------------------------------------------------------------------
extern "C" void kernel_launch(void* const* d_in, const int* in_sizes, int n_in,
                              void* d_out, int out_size, void* d_ws, size_t ws_size,
                              hipStream_t stream) {
  const float* q    = (const float*)d_in[0];
  const float* k    = (const float*)d_in[1];
  const float* v    = (const float*)d_in[2];
  const int*   mask = (const int*)d_in[3];
  const float* Wq   = (const float*)d_in[4];
  const float* bq   = (const float*)d_in[5];
  const float* Wk   = (const float*)d_in[6];
  const float* bk   = (const float*)d_in[7];
  const float* Wv   = (const float*)d_in[8];
  const float* bv   = (const float*)d_in[9];
  const float* Wo   = (const float*)d_in[10];
  const float* bo   = (const float*)d_in[11];

  const size_t HT = (size_t)NBATCH * HN * LSEQ * DKH;  // 4.19M halves each
  _Float16* Q16 = (_Float16*)d_ws;
  _Float16* K16 = Q16 + HT;
  _Float16* V16 = K16 + HT;
  _Float16* C16 = V16 + HT;   // total 32 MB of workspace

  dim3 gg((NBATCH * LSEQ) / 128, DMODEL / 64);   // (64, 8)
  // Q projection pre-scaled by 1/sqrt(64) (exact power of two).
  gemm_xwt<float, false><<<gg, 256, 0, stream>>>(q, Wq, bq, Q16, 0.125f);
  gemm_xwt<float, false><<<gg, 256, 0, stream>>>(k, Wk, bk, K16, 1.0f);
  gemm_xwt<float, false><<<gg, 256, 0, stream>>>(v, Wv, bv, V16, 1.0f);

  dim3 ga(LSEQ / 128, NBATCH * HN);              // (16, 32)
  attn_fa<<<ga, 256, 0, stream>>>(Q16, K16, V16, mask, C16);

  gemm_xwt<_Float16, true><<<gg, 256, 0, stream>>>(C16, Wo, bo, d_out, 1.0f);
}
